// GRU_69501160784000
// MI455X (gfx1250) — compile-verified, matrix operand order checked by reading, not run on hardware
//
#include <hip/hip_runtime.h>
#include <math.h>

// ---- problem constants -------------------------------------------------
constexpr int kT  = 512;
constexpr int kB  = 64;
constexpr int kIC = 512;
constexpr int kHC = 512;
constexpr int kL  = 2;

// ---- CDNA5 WMMA types --------------------------------------------------
typedef __bf16 bf16_t;
typedef bf16_t v16bf __attribute__((ext_vector_type(16)));
typedef bf16_t v8bf  __attribute__((ext_vector_type(8)));
typedef float  v8f   __attribute__((ext_vector_type(8)));

__device__ __forceinline__ bf16_t f2bf(float f) {
  unsigned u = __float_as_uint(f);
  unsigned r = (u + 0x7FFFu + ((u >> 16) & 1u)) >> 16;   // round-to-nearest-even
  unsigned short h = (unsigned short)r;
  return __builtin_bit_cast(bf16_t, h);
}

// 16x32 bf16 fragment (A-layout; also B = W^T by reading W's rows).
// ISA 7.12.2: lane<16: elems 0..7 = K k0..+7, 8..15 = K k0+16..+23
//             lane>=16: elems 0..7 = K k0+8..+15, 8..15 = K k0+24..+31
template <typename P>
__device__ __forceinline__ v16bf load_frag(P base, int ld, int row0, int k0, int lane) {
  int m  = lane & 15;
  int kh = lane >> 4;
  auto p = base + (size_t)(row0 + m) * ld + (k0 + kh * 8);
  v8bf lo = *(const v8bf*)p;
  v8bf hi = *(const v8bf*)(p + 16);
  v16bf r;
#pragma unroll
  for (int i = 0; i < 8; i++) { r[i] = lo[i]; r[i + 8] = hi[i]; }
  return r;
}

// ---- conversion kernels ------------------------------------------------
__global__ void convert_x_kernel(const float* __restrict__ src, bf16_t* __restrict__ dst, int n) {
  int i = blockIdx.x * blockDim.x + threadIdx.x;
  if (i < n) dst[i] = f2bf(src[i]);
}

// Per-layer WBF layout (elems): W1x @0, W1h @524288, W2x @1048576, W2h @1310720.
__global__ void convert_w_kernel(const float* __restrict__ W1, const float* __restrict__ W2,
                                 bf16_t* __restrict__ WBF) {
  int idx = blockIdx.x * blockDim.x + threadIdx.x;
  const int perLayer = 1536 * 1024;
  if (idx >= 2 * perLayer) return;
  int l = idx / perLayer;
  int rem = idx % perLayer;
  bf16_t* base = WBF + (size_t)l * 1572864;
  if (rem < 1024 * 1024) {
    int r = rem >> 10, c = rem & 1023;
    bf16_t v = f2bf(W1[(size_t)l * 1048576 + rem]);
    if (c < 512) base[r * 512 + c] = v;
    else         base[524288 + r * 512 + (c - 512)] = v;
  } else {
    int rem2 = rem - 1048576;
    int r = rem2 >> 10, c = rem2 & 1023;
    bf16_t v = f2bf(W2[(size_t)l * 524288 + rem2]);
    if (c < 512) base[1048576 + r * 512 + c] = v;
    else         base[1310720 + r * 512 + (c - 512)] = v;
  }
}

__global__ void init_layer_kernel(const float* __restrict__ h0, float* __restrict__ H32,
                                  bf16_t* __restrict__ HBF, unsigned* __restrict__ BAR) {
  int i = blockIdx.x * blockDim.x + threadIdx.x;
  if (i < kB * kHC) {
    float v = h0[i & (kHC - 1)];
    H32[i] = v;
    HBF[i] = f2bf(v);
  }
  if (i < 8) BAR[i] = 0;
}

// ---- phase 1: input projections, all timesteps -------------------------
// One block per M-tile. A-slice (16x512 bf16 = 16KB) staged to LDS with
// gfx1250 async-to-LDS DMA, held in registers, reused across 12 N-tiles per
// wave.  N-tiles processed in pairs: two independent accumulator chains let
// WMMAs of one tile cover the other's B-tile load latency.
__global__ void __launch_bounds__(256)
xproj_kernel(const bf16_t* __restrict__ in, const bf16_t* __restrict__ W1x,
             const bf16_t* __restrict__ W2x, const float* __restrict__ b1l,
             const float* __restrict__ b2l, float* __restrict__ XPRE) {
  __shared__ bf16_t As[16 * 512];                 // 16KB A slice
  const int tid = threadIdx.x;
  const int mt = blockIdx.x;                      // 2048 M-tiles
  const int row0 = mt * 16;

  // --- async DMA: global -> LDS (4 x b128 per lane, ASYNCcnt tracked) ---
  {
    const char* src = (const char*)(in + (size_t)row0 * 512);
    unsigned lds_base = (unsigned)(size_t)(&As[0]);
#pragma unroll
    for (int it = 0; it < 4; it++) {
      unsigned off = (unsigned)((it * 256 + tid) * 16);
      unsigned lds = lds_base + off;
      unsigned long long ga = (unsigned long long)(size_t)src + off;
      asm volatile("global_load_async_to_lds_b128 %0, %1, off"
                   :: "v"(lds), "v"(ga) : "memory");
    }
    asm volatile("s_wait_asynccnt 0" ::: "memory");
  }
  __syncthreads();

  const int wid = tid >> 5, lane = tid & 31;
  const int n = lane & 15, kh = lane >> 4;

  // A fragments: loaded from LDS once, reused for 12 N-tiles (128 VGPRs).
  v16bf afrag[16];
#pragma unroll
  for (int kk = 0; kk < 16; kk++)
    afrag[kk] = load_frag((const bf16_t*)As, 512, 0, kk * 32, lane);

  for (int j = 0; j < 6; j++) {
    int nt0 = wid * 12 + 2 * j;                   // even, pair never straddles 63|64
    int nt1 = nt0 + 1;

    const bf16_t *Wp0, *Wp1;
    float bv0, bv1;
    int oc0, oc1;
    if (nt0 < 64) { Wp0 = W1x + (size_t)nt0 * 8192; bv0 = b1l[nt0 * 16 + n]; oc0 = nt0 * 16; }
    else { int q = nt0 - 64; Wp0 = W2x + (size_t)q * 8192; bv0 = b2l[q * 16 + n]; oc0 = 1024 + q * 16; }
    if (nt1 < 64) { Wp1 = W1x + (size_t)nt1 * 8192; bv1 = b1l[nt1 * 16 + n]; oc1 = nt1 * 16; }
    else { int q = nt1 - 64; Wp1 = W2x + (size_t)q * 8192; bv1 = b2l[q * 16 + n]; oc1 = 1024 + q * 16; }

    v8f acc0, acc1;
#pragma unroll
    for (int r = 0; r < 8; r++) { acc0[r] = bv0; acc1[r] = bv1; }

#pragma unroll
    for (int kk = 0; kk < 16; kk++) {
      v16bf b0 = load_frag(Wp0, 512, 0, kk * 32, lane);
      v16bf b1 = load_frag(Wp1, 512, 0, kk * 32, lane);
      acc0 = __builtin_amdgcn_wmma_f32_16x16x32_bf16(false, afrag[kk], false, b0, (short)0, acc0, false, false);
      acc1 = __builtin_amdgcn_wmma_f32_16x16x32_bf16(false, afrag[kk], false, b1, (short)0, acc1, false, false);
    }

    float* outp = XPRE + (size_t)row0 * 1536;
#pragma unroll
    for (int r = 0; r < 8; r++) {
      outp[(size_t)(r + kh * 8) * 1536 + oc0 + n] = acc0[r];
      outp[(size_t)(r + kh * 8) * 1536 + oc1 + n] = acc1[r];
    }
  }
}

// ---- grid-wide barrier (persistent cooperative kernel) -----------------
__device__ __forceinline__ void grid_barrier(unsigned* bar, unsigned nblocks) {
  __syncthreads();
  if (threadIdx.x == 0) {
    __threadfence();
    volatile unsigned* vgen = bar + 1;
    unsigned gen = *vgen;
    unsigned prev = atomicAdd(bar, 1u);
    if (prev == nblocks - 1) {
      bar[0] = 0u;
      __threadfence();
      atomicAdd(bar + 1, 1u);
    } else {
      while (*vgen == gen) { __builtin_amdgcn_s_sleep(1); }
    }
    __threadfence();
  }
  __syncthreads();
}

// ---- phase 2: sequential recurrence (48 persistent blocks) -------------
// Blocks 0..31 (256 waves, "A"): pre = XPRE1[t] + h @ W1h^T; sigmoid; emit r*h, z.
// Blocks 32..47 (128 waves, "B"): g = tanh(XPRE2[t] + (r*h) @ W2h^T); h update.
// Weight tile lives in 128 VGPRs for the whole kernel; per step all 16
// h-fragments are loaded as one clause (32 b128 in flight) so the serial
// path pays ~one L2 round-trip per step instead of sixteen.
__global__ void __launch_bounds__(256)
recurrent_kernel(int layer, const float* __restrict__ XPRE,
                 const bf16_t* __restrict__ W1h, const bf16_t* __restrict__ W2h,
                 float* __restrict__ H32, bf16_t* __restrict__ HBF,
                 float* __restrict__ ZBUF, bf16_t* __restrict__ RHBF,
                 bf16_t* __restrict__ OUT1BF, float* __restrict__ outF,
                 unsigned* __restrict__ BAR) {
  const unsigned NB = gridDim.x;                  // 48
  const int lane = threadIdx.x & 31;
  const int gw = blockIdx.x * (blockDim.x >> 5) + (threadIdx.x >> 5);  // 0..383
  const int n = lane & 15, kh = lane >> 4;
  const bool isA = (gw < 256);

  int mt, col0;
  const bf16_t* Wp;
  if (isA) { mt = gw >> 6; col0 = (gw & 63) * 16; Wp = W1h + (size_t)col0 * 512; }
  else { int gb = gw - 256; mt = gb >> 5; col0 = (gb & 31) * 16; Wp = W2h + (size_t)col0 * 512; }
  const int row0 = mt * 16;

  // loop-invariant weight tile -> registers (16 x v16bf = 128 VGPRs)
  v16bf wfrag[16];
#pragma unroll
  for (int kk = 0; kk < 16; kk++) wfrag[kk] = load_frag(Wp, 512, 0, kk * 32, lane);

  for (int t = 0; t < kT; t++) {
    const float* xp = XPRE + (size_t)t * (kB * 1536);

    if (isA) {
      // ---- Phase A: gates GEMM + sigmoid, fused r*h ----
      v8f acc;
#pragma unroll
      for (int r = 0; r < 8; r++)
        acc[r] = xp[(size_t)(row0 + r + kh * 8) * 1536 + col0 + n];

      v16bf a[16];                                // all h-frag loads in flight
#pragma unroll
      for (int kk = 0; kk < 16; kk++) a[kk] = load_frag(HBF, 512, row0, kk * 32, lane);
#pragma unroll
      for (int kk = 0; kk < 16; kk++)
        acc = __builtin_amdgcn_wmma_f32_16x16x32_bf16(false, a[kk], false, wfrag[kk], (short)0, acc, false, false);

#pragma unroll
      for (int r = 0; r < 8; r++) {
        int m = row0 + r + kh * 8;
        int col = col0 + n;
        float s = 1.0f / (1.0f + __expf(-acc[r]));
        if (col < kHC) {                          // wave-uniform branch
          float rh = s * H32[m * kHC + col];
          RHBF[m * kHC + col] = f2bf(rh);
        } else {
          ZBUF[m * kHC + (col - kHC)] = s;
        }
      }
    } else {
      // B-waves idle: prefetch their XPRE2[t] accumulator tile for phase B.
#pragma unroll
      for (int r = 0; r < 8; r++)
        __builtin_prefetch(&xp[(size_t)(row0 + r + kh * 8) * 1536 + 1024 + col0 + n], 0, 3);
    }
    grid_barrier(BAR, NB);

    if (!isA) {
      // ---- Phase B: candidate GEMM + tanh + h update ----
      v8f acc;
#pragma unroll
      for (int r = 0; r < 8; r++)
        acc[r] = xp[(size_t)(row0 + r + kh * 8) * 1536 + 1024 + col0 + n];

      v16bf a[16];
#pragma unroll
      for (int kk = 0; kk < 16; kk++) a[kk] = load_frag(RHBF, 512, row0, kk * 32, lane);
#pragma unroll
      for (int kk = 0; kk < 16; kk++)
        acc = __builtin_amdgcn_wmma_f32_16x16x32_bf16(false, a[kk], false, wfrag[kk], (short)0, acc, false, false);

#pragma unroll
      for (int r = 0; r < 8; r++) {
        int m = row0 + r + kh * 8;
        int col = col0 + n;
        float g = tanhf(acc[r]);
        float z = ZBUF[m * kHC + col];
        float ho = H32[m * kHC + col];
        float hn = z * ho + (1.0f - z) * g;
        H32[m * kHC + col] = hn;
        HBF[m * kHC + col] = f2bf(hn);
        size_t orow = (size_t)t * kB + m;
        if (layer == 0) OUT1BF[orow * kHC + col] = f2bf(hn);
        else            outF[orow * kHC + col] = hn;
        if (t == kT - 1)
          outF[(size_t)kT * kB * kHC + (size_t)layer * kB * kHC + m * kHC + col] = hn;
      }
    } else if (t + 1 < kT) {
      // A-waves idle: prefetch their XPRE1[t+1] accumulator tile.
      const float* xpn = XPRE + (size_t)(t + 1) * (kB * 1536);
#pragma unroll
      for (int r = 0; r < 8; r++)
        __builtin_prefetch(&xpn[(size_t)(row0 + r + kh * 8) * 1536 + col0 + n], 0, 3);
    }
    grid_barrier(BAR, NB);
  }
}

// ---- host launcher -----------------------------------------------------
extern "C" void kernel_launch(void* const* d_in, const int* in_sizes, int n_in,
                              void* d_out, int out_size, void* d_ws, size_t ws_size,
                              hipStream_t stream) {
  const float* x       = (const float*)d_in[0];   // [T,B,IC]
  const float* hiddens = (const float*)d_in[1];   // [L,1,HC]
  const float* W1      = (const float*)d_in[2];   // [L,2HC,IC+HC]
  const float* b1      = (const float*)d_in[3];   // [L,2HC]
  const float* W2      = (const float*)d_in[4];   // [L,HC,IC+HC]
  const float* b2      = (const float*)d_in[5];   // [L,HC]
  float* outF = (float*)d_out;                    // out [T,B,HC] ++ hs [L,B,HC]

  char* ws = (char*)d_ws;
  size_t o = 0;
  bf16_t* XBF    = (bf16_t*)(ws + o); o += (size_t)kT * kB * kIC * 2;     // 32 MB
  bf16_t* OUT1BF = (bf16_t*)(ws + o); o += (size_t)kT * kB * kHC * 2;     // 32 MB
  bf16_t* WBF    = (bf16_t*)(ws + o); o += (size_t)2 * 1572864 * 2;       // 6 MB
  float*  XPRE   = (float*)(ws + o);  o += (size_t)kT * kB * 1536 * 4;    // 192 MB
  float*  H32    = (float*)(ws + o);  o += (size_t)kB * kHC * 4;
  float*  ZBUF   = (float*)(ws + o);  o += (size_t)kB * kHC * 4;
  bf16_t* HBF    = (bf16_t*)(ws + o); o += (size_t)kB * kHC * 2;
  bf16_t* RHBF   = (bf16_t*)(ws + o); o += (size_t)kB * kHC * 2;
  unsigned* BAR  = (unsigned*)(ws + o); o += 256;

  const int nX = kT * kB * kIC;
  convert_x_kernel<<<nX / 256, 256, 0, stream>>>(x, XBF, nX);
  convert_w_kernel<<<(2 * 1536 * 1024) / 256, 256, 0, stream>>>(W1, W2, WBF);

  for (int l = 0; l < kL; l++) {
    bf16_t* W1x_l = WBF + (size_t)l * 1572864;
    bf16_t* W1h_l = W1x_l + 524288;
    bf16_t* W2x_l = W1x_l + 1048576;
    bf16_t* W2h_l = W1x_l + 1310720;

    init_layer_kernel<<<(kB * kHC) / 256, 256, 0, stream>>>(hiddens + l * kHC, H32, HBF, BAR);

    const bf16_t* layer_in = (l == 0) ? XBF : OUT1BF;
    // one block per M-tile; 8 waves x 12 N-tiles (pairs) cover [W1x | W2x]
    xproj_kernel<<<2048, 256, 0, stream>>>(layer_in, W1x_l, W2x_l,
                                           b1 + (size_t)l * 2 * kHC,
                                           b2 + (size_t)l * kHC, XPRE);

    recurrent_kernel<<<48, 256, 0, stream>>>(l, XPRE, W1h_l, W2h_l,
                                             H32, HBF, ZBUF, RHBF,
                                             OUT1BF, outF, BAR);
  }
}
